// GGNN_2087354106026
// MI455X (gfx1250) — matrix operand order repaired and checked by reading, not compiled, not run on hardware
//
#include <hip/hip_runtime.h>
#include <hip/hip_bf16.h>

// ---------------------------------------------------------------------------
// SR-GNN (GGNN step=1 + attention readout + full-vocab scoring) for gfx1250.
// All GEMMs run through v_wmma_f32_16x16x32_f16 with f32 accumulation;
// elementwise / gather stages are pure-bandwidth kernels.
// Staging is branch-free: clamped-address b128 loads + cndmask zeroing, so the
// memory pipeline keeps multiple loads in flight (no per-element waits).
// ---------------------------------------------------------------------------

typedef _Float16 h4v  __attribute__((ext_vector_type(4)));
typedef _Float16 h8v  __attribute__((ext_vector_type(8)));
typedef _Float16 v16h __attribute__((ext_vector_type(16)));
typedef float    f4v  __attribute__((ext_vector_type(4)));
typedef float    v8f  __attribute__((ext_vector_type(8)));

union AFrag { v16h v; h8v h[2]; };

__device__ __forceinline__ float sigmoidf_(float x) {
  return 1.0f / (1.0f + __expf(-x));
}
__device__ __forceinline__ int imin_(int a, int b) { return a < b ? a : b; }

// ---------------------------------------------------------------------------
// Generic tiled WMMA GEMM:  C = act(A[MxK] * B[KxN] + bias)
//   A: row-major f32 (lda). B: row-major f32, KxN (BTRANS=0, ldb=N-stride) or
//   NxK (BTRANS=1, ldb=K-stride; logits = y1 @ embedding[1:]^T). C: f32 (ldc).
// 128 threads = 4 waves; wave owns a 16x64 strip (4 accumulators); K staged
// 32-deep in f16 LDS with zero padding (K and non-trans N must be %4==0,
// true for all call sites: K in {100,200,300}, N in {100,200}).
//
// Fragment layouts per CDNA5 ISA 7.12.2:
//   A (16-bit 16x32): lane l = row M=l%16; two contiguous 8-half K segments
//     [lh*8..lh*8+7] and [16+lh*8..16+lh*8+7], lh = l/16 -> 2x ds_load_b128.
//   B (32x16): lane l = column N=l%16; same K segments as A -> 2x b128.
//   D (f32 16x16): VGPR r holds M = r + lh*8, N = l%16.
// ---------------------------------------------------------------------------
template <int ACT, int BTRANS>
__global__ __launch_bounds__(128) void wmma_gemm(
    const float* __restrict__ A, const float* __restrict__ Bsrc,
    const float* __restrict__ bias, float* __restrict__ C,
    int M, int N, int K, int lda, int ldb, int ldc) {
  __shared__ __align__(16) _Float16 Ash[64][32];
  __shared__ __align__(16) _Float16 Bsh[64][32];  // [n][k]

  const int tid  = threadIdx.x;
  const int wave = tid >> 5;
  const int lane = tid & 31;
  const int lh   = lane >> 4;  // 0 or 1
  const int l16  = lane & 15;

  const int m0 = blockIdx.y * 64;
  const int n0 = blockIdx.x * 64;

  // staging coordinates (fixed per thread)
  const int arow = tid >> 3;            // 0..15 (A / B-trans tiles)
  const int akq  = (tid & 7) * 4;       // 0,4,...,28
  const int bkk  = tid >> 4;            // 0..7  (B K x N tiles)
  const int bnq  = (tid & 15) * 4;      // 0,4,...,60

  const f4v fzero = {0.0f, 0.0f, 0.0f, 0.0f};
  v8f acc[4] = {};

  for (int k0 = 0; k0 < K; k0 += 32) {
    // ---- stage A tile (64 x 32): 4 unconditional b128 loads, cndmask-zero --
#pragma unroll
    for (int it = 0; it < 4; ++it) {
      const int r  = it * 16 + arow;
      const int gr = m0 + r, gk = k0 + akq;
      const bool ok = (gr < M) & (gk < K);
      const float* p = ok ? (A + (size_t)gr * lda + gk) : A;
      f4v v = *(const f4v*)p;
      v = ok ? v : fzero;
      h4v hv = {(_Float16)v[0], (_Float16)v[1], (_Float16)v[2], (_Float16)v[3]};
      *(h4v*)&Ash[r][akq] = hv;
    }
    // ---- stage B tile into [n][k] f16 layout ----
    if (BTRANS) {
#pragma unroll
      for (int it = 0; it < 4; ++it) {
        const int n  = it * 16 + arow;
        const int gn = n0 + n, gk = k0 + akq;
        const bool ok = (gn < N) & (gk < K);
        const float* p = ok ? (Bsrc + (size_t)gn * ldb + gk) : Bsrc;
        f4v v = *(const f4v*)p;
        v = ok ? v : fzero;
        h4v hv = {(_Float16)v[0], (_Float16)v[1], (_Float16)v[2], (_Float16)v[3]};
        *(h4v*)&Bsh[n][akq] = hv;
      }
    } else {
#pragma unroll
      for (int it = 0; it < 4; ++it) {
        const int kk = it * 8 + bkk;
        const int gk = k0 + kk, gn = n0 + bnq;
        const bool ok = (gk < K) & (gn < N);
        const float* p = ok ? (Bsrc + (size_t)gk * ldb + gn) : Bsrc;
        f4v v = *(const f4v*)p;
        v = ok ? v : fzero;
        Bsh[bnq + 0][kk] = (_Float16)v[0];
        Bsh[bnq + 1][kk] = (_Float16)v[1];
        Bsh[bnq + 2][kk] = (_Float16)v[2];
        Bsh[bnq + 3][kk] = (_Float16)v[3];
      }
    }
    // unconditional clamped prefetch of next K tile (global_prefetch_b8)
    if (k0 + 32 < K) {
      const int pr = imin_(m0 + (tid & 63), M - 1);
      __builtin_prefetch(A + (size_t)pr * lda + (k0 + 32), 0, 1);
      if (BTRANS) {
        const int pn = imin_(n0 + (tid & 63), N - 1);
        __builtin_prefetch(Bsrc + (size_t)pn * ldb + (k0 + 32), 0, 1);
      } else {
        const int pk = imin_(k0 + 32 + (tid & 31), K - 1);
        __builtin_prefetch(Bsrc + (size_t)pk * ldb + n0, 0, 1);
      }
    }
    __syncthreads();

    // ---- A fragment for this wave's 16-row strip ----
    AFrag af;
    const int ar = wave * 16 + l16;
    af.h[0] = *(const h8v*)&Ash[ar][lh * 8];
    af.h[1] = *(const h8v*)&Ash[ar][16 + lh * 8];

    // ---- 4 WMMAs across the 64-wide N strip ----
#pragma unroll
    for (int ns = 0; ns < 4; ++ns) {
      AFrag bf;
      const int bc = ns * 16 + l16;
      bf.h[0] = *(const h8v*)&Bsh[bc][lh * 8];
      bf.h[1] = *(const h8v*)&Bsh[bc][16 + lh * 8];
      acc[ns] = __builtin_amdgcn_wmma_f32_16x16x32_f16(
          false, af.v, false, bf.v, (short)0, acc[ns], false, false);
    }
    __syncthreads();
  }

  // ---- epilogue: bias + activation, predicated store ----
#pragma unroll
  for (int ns = 0; ns < 4; ++ns) {
    const int col = n0 + ns * 16 + l16;
    const float bv = (bias != nullptr && col < N) ? bias[col] : 0.0f;
#pragma unroll
    for (int r = 0; r < 8; ++r) {
      const int row = m0 + wave * 16 + r + lh * 8;
      if (row < M && col < N) {
        float v = acc[ns][r] + bv;
        if (ACT == 1) v = sigmoidf_(v);
        if (ACT == 2) v = tanhf(v);
        C[(size_t)row * ldc + col] = v;
      }
    }
  }
}

// ---------------------------------------------------------------------------
// Bandwidth-bound helper kernels
// ---------------------------------------------------------------------------

// h[bt, :] = embedding[item[bt], :]
__global__ void k_gather_h(const float* __restrict__ emb,
                           const int* __restrict__ item,
                           float* __restrict__ h, int total, int d) {
  int i = blockIdx.x * blockDim.x + threadIdx.x;
  if (i >= total) return;
  int bt = i / d, j = i - bt * d;
  h[i] = emb[(size_t)item[bt] * d + j];
}

// Per-batch: a_in = adj_in(50x50) @ hin(50x100); a_out likewise.
// X[bt] = [a_in | a_out | h]  (300 cols)
__global__ __launch_bounds__(256) void k_adj_concat(
    const float* __restrict__ adj_in, const float* __restrict__ adj_out,
    const float* __restrict__ hin, const float* __restrict__ hout,
    const float* __restrict__ h, float* __restrict__ X, int T, int d) {
  const int b = blockIdx.x;
  __shared__ float sin_[50 * 100];
  __shared__ float sout_[50 * 100];
  for (int i = threadIdx.x; i < T * d; i += blockDim.x) {
    sin_[i]  = hin[(size_t)b * T * d + i];
    sout_[i] = hout[(size_t)b * T * d + i];
  }
  __syncthreads();
  for (int i = threadIdx.x; i < T * d; i += blockDim.x) {
    int t = i / d, j = i - t * d;
    const float* Ai = adj_in  + ((size_t)b * T + t) * T;
    const float* Ao = adj_out + ((size_t)b * T + t) * T;
    float ain = 0.0f, aout = 0.0f;
    for (int s = 0; s < T; ++s) {
      ain  += Ai[s] * sin_[s * d + j];
      aout += Ao[s] * sout_[s * d + j];
    }
    size_t bt = (size_t)b * T + t;
    X[bt * 300 + j]       = ain;
    X[bt * 300 + 100 + j] = aout;
    X[bt * 300 + 200 + j] = h[bt * d + j];
  }
}

// X[:, 200:300] = r * h  (r = gates[:, 0:100])
__global__ void k_rh(const float* __restrict__ gates,
                     const float* __restrict__ h, float* __restrict__ X,
                     int total, int d) {
  int i = blockIdx.x * blockDim.x + threadIdx.x;
  if (i >= total) return;
  int bt = i / d, j = i - bt * d;
  float r = gates[(size_t)bt * 2 * d + j];
  X[(size_t)bt * 3 * d + 2 * d + j] = r * h[i];
}

// h' = u*h + (1-u)*c   (u = gates[:, 100:200])
__global__ void k_hnew(const float* __restrict__ gates,
                       const float* __restrict__ h,
                       const float* __restrict__ cand,
                       float* __restrict__ reemb, int total, int d) {
  int i = blockIdx.x * blockDim.x + threadIdx.x;
  if (i >= total) return;
  int bt = i / d, j = i - bt * d;
  float u = gates[(size_t)bt * 2 * d + d + j];
  reemb[i] = u * h[i] + (1.0f - u) * cand[i];
}

// seq_h gather via alias; last_h via alias[b, sum(mask)-1]
__global__ __launch_bounds__(128) void k_gather_seq(
    const float* __restrict__ reemb, const int* __restrict__ alias,
    const float* __restrict__ mask, float* __restrict__ seqh,
    float* __restrict__ lasth, int T, int d) {
  const int b = blockIdx.x;
  __shared__ float red[128];
  float s = 0.0f;
  for (int t = threadIdx.x; t < T; t += blockDim.x) s += mask[(size_t)b * T + t];
  red[threadIdx.x] = s;
  __syncthreads();
  for (int st = 64; st > 0; st >>= 1) {
    if (threadIdx.x < st) red[threadIdx.x] += red[threadIdx.x + st];
    __syncthreads();
  }
  const int rm = (int)red[0];
  const int last_id = alias[(size_t)b * T + (rm - 1)];
  for (int i = threadIdx.x; i < T * d; i += blockDim.x) {
    int t = i / d, j = i - t * d;
    int a = alias[(size_t)b * T + t];
    seqh[((size_t)b * T + t) * d + j] = reemb[((size_t)b * T + a) * d + j];
  }
  for (int j = threadIdx.x; j < d; j += blockDim.x)
    lasth[(size_t)b * d + j] = reemb[((size_t)b * T + last_id) * d + j];
}

// coef[bt] = ( sum_j sigmoid(last[b]+seq[bt]+nasr_b)_j * nasr_v_j ) * mask[bt]
__global__ __launch_bounds__(128) void k_coef(
    const float* __restrict__ last, const float* __restrict__ seq,
    const float* __restrict__ nasr_b, const float* __restrict__ nasr_v,
    const float* __restrict__ mask, float* __restrict__ coef, int T, int d) {
  const int bt = blockIdx.x;
  const int b = bt / T;
  float p = 0.0f;
  for (int j = threadIdx.x; j < d; j += blockDim.x) {
    float x = last[(size_t)b * d + j] + seq[(size_t)bt * d + j] + nasr_b[j];
    p += sigmoidf_(x) * nasr_v[j];
  }
  __shared__ float red[128];
  red[threadIdx.x] = p;
  __syncthreads();
  for (int st = 64; st > 0; st >>= 1) {
    if (threadIdx.x < st) red[threadIdx.x] += red[threadIdx.x + st];
    __syncthreads();
  }
  if (threadIdx.x == 0) coef[bt] = red[0] * mask[bt];
}

// ma[b] = [ sum_t coef[b,t]*seq_h[b,t,:]  |  last[b,:] ]
__global__ __launch_bounds__(128) void k_ma(
    const float* __restrict__ coef, const float* __restrict__ seqh,
    const float* __restrict__ last, float* __restrict__ ma, int T, int d) {
  const int b = blockIdx.x;
  for (int j = threadIdx.x; j < d; j += blockDim.x) {
    float s = 0.0f;
    for (int t = 0; t < T; ++t)
      s += coef[(size_t)b * T + t] * seqh[((size_t)b * T + t) * d + j];
    ma[(size_t)b * 2 * d + j]     = s;
    ma[(size_t)b * 2 * d + d + j] = last[(size_t)b * d + j];
  }
}

// ---------------------------------------------------------------------------
// Host launcher
// ---------------------------------------------------------------------------
extern "C" void kernel_launch(void* const* d_in, const int* in_sizes, int n_in,
                              void* d_out, int out_size, void* d_ws,
                              size_t ws_size, hipStream_t stream) {
  (void)in_sizes; (void)n_in; (void)out_size; (void)ws_size;
  constexpr int d = 100, T = 50, Bb = 512, BT = Bb * T;  // 25600
  constexpr int V = 99999;  // logits columns = N_NODE - 1

  const float* adj_in  = (const float*)d_in[0];
  const float* adj_out = (const float*)d_in[1];
  const float* mask    = (const float*)d_in[2];
  const int*   item    = (const int*)d_in[3];
  const int*   alias   = (const int*)d_in[4];
  // d_in[5] = step (device scalar; setup fixes step=1, honored at compile time)
  const float* emb     = (const float*)d_in[6];
  const float* W_in    = (const float*)d_in[7];
  const float* b_in    = (const float*)d_in[8];
  const float* W_out   = (const float*)d_in[9];
  const float* b_out   = (const float*)d_in[10];
  const float* gate_k  = (const float*)d_in[11];
  const float* gate_b  = (const float*)d_in[12];
  const float* cand_k  = (const float*)d_in[13];
  const float* cand_b  = (const float*)d_in[14];
  const float* nasr_w1 = (const float*)d_in[15];
  const float* nasr_w2 = (const float*)d_in[16];
  const float* nasr_v  = (const float*)d_in[17];
  const float* nasr_b  = (const float*)d_in[18];
  const float* B_mat   = (const float*)d_in[19];
  float* out = (float*)d_out;

  // workspace layout (bytes, 256-aligned); explicit buffer reuse
  char* w = (char*)d_ws;
  float* h_buf = (float*)(w + 0);           // BT*100  (reused as seqh)
  float* X     = (float*)(w + 10240000);    // BT*300
  float* hin   = (float*)(w + 40960000);    // BT*100  (reused as cand)
  float* hout  = (float*)(w + 51200000);    // BT*100  (reused as reemb)
  float* gates = (float*)(w + 61440000);    // BT*200  (reused as seq)
  float* lasth = (float*)(w + 81920000);    // 512*100
  float* last  = (float*)(w + 82124800);    // 512*100
  float* coef  = (float*)(w + 82329600);    // BT
  float* ma    = (float*)(w + 82432000);    // 512*200
  float* y1    = (float*)(w + 82841600);    // 512*100
  float* cand  = hin;    // hin dead after k_adj_concat
  float* reemb = hout;   // hout dead after k_adj_concat
  float* seqh  = h_buf;  // h dead after k_hnew
  float* seq   = gates;  // gates dead after k_hnew

  auto ggrid = [](int M, int N) { return dim3((N + 63) / 64, (M + 63) / 64, 1); };
  const int totBTd = BT * d;
  const int ewBlocks = (totBTd + 255) / 256;

  // 1. gather h = embedding[item]
  k_gather_h<<<ewBlocks, 256, 0, stream>>>(emb, item, h_buf, totBTd, d);

  // 2. hin = h@W_in + b_in ; hout = h@W_out + b_out   (WMMA)
  wmma_gemm<0, 0><<<ggrid(BT, d), 128, 0, stream>>>(h_buf, W_in, b_in, hin,
                                                    BT, d, d, d, d, d);
  wmma_gemm<0, 0><<<ggrid(BT, d), 128, 0, stream>>>(h_buf, W_out, b_out, hout,
                                                    BT, d, d, d, d, d);

  // 3. X = [adj_in@hin | adj_out@hout | h]
  k_adj_concat<<<Bb, 256, 0, stream>>>(adj_in, adj_out, hin, hout, h_buf, X, T, d);

  // 4. gates = sigmoid(X @ gru_gate_k + gru_gate_b)   (WMMA, K=300 padded)
  wmma_gemm<1, 0><<<ggrid(BT, 2 * d), 128, 0, stream>>>(
      X, gate_k, gate_b, gates, BT, 2 * d, 3 * d, 3 * d, 2 * d, 2 * d);

  // 5. X[:,200:300] = r * h
  k_rh<<<ewBlocks, 256, 0, stream>>>(gates, h_buf, X, totBTd, d);

  // 6. cand = tanh(X @ gru_cand_k + gru_cand_b)       (WMMA)
  wmma_gemm<2, 0><<<ggrid(BT, d), 128, 0, stream>>>(
      X, cand_k, cand_b, cand, BT, d, 3 * d, 3 * d, d, d);

  // 7. reemb = u*h + (1-u)*cand
  k_hnew<<<ewBlocks, 256, 0, stream>>>(gates, h_buf, cand, reemb, totBTd, d);

  // 8. seq_h / last_h gathers via alias + mask-sum
  k_gather_seq<<<Bb, 128, 0, stream>>>(reemb, alias, mask, seqh, lasth, T, d);

  // 9. seq = seq_h @ nasr_w2 ; last = last_h @ nasr_w1   (WMMA)
  wmma_gemm<0, 0><<<ggrid(BT, d), 128, 0, stream>>>(seqh, nasr_w2, nullptr, seq,
                                                    BT, d, d, d, d, d);
  wmma_gemm<0, 0><<<ggrid(Bb, d), 128, 0, stream>>>(lasth, nasr_w1, nullptr, last,
                                                    Bb, d, d, d, d, d);

  // 10-11. attention coefficients and readout ma
  k_coef<<<BT, 128, 0, stream>>>(last, seq, nasr_b, nasr_v, mask, coef, T, d);
  k_ma<<<Bb, 128, 0, stream>>>(coef, seqh, last, ma, T, d);

  // 12. y1 = ma @ B_mat   (WMMA, K=200)
  wmma_gemm<0, 0><<<ggrid(Bb, d), 128, 0, stream>>>(ma, B_mat, nullptr, y1,
                                                    Bb, d, 2 * d, 2 * d, d, d);

  // 13. logits = y1 @ embedding[1:]^T   (WMMA, B-transposed path; dominant GEMM)
  wmma_gemm<0, 1><<<ggrid(Bb, V), 128, 0, stream>>>(y1, emb + d, nullptr, out,
                                                    Bb, V, d, d, d, V);
}
